// Recommender_70841190580317
// MI455X (gfx1250) — matrix operand order repaired and checked by reading, not compiled, not run on hardware
//
#include <hip/hip_runtime.h>
#include <hip/hip_bf16.h>

typedef __attribute__((ext_vector_type(16))) _Float16 v16h;
typedef __attribute__((ext_vector_type(8)))  _Float16 v8h;
typedef __attribute__((ext_vector_type(8)))  float    v8f;

#define N_VENUES 100000
#define N_ROWS_A 1856   /* 64*29 rows for loss */
#define N_ROWS_B 192    /* 64*3  rows for top-k */
#define NV_TILE  512
#define NSUB     (NV_TILE/16)

enum { MODE_MAX = 0, MODE_Z = 1, MODE_S2 = 2, MODE_RANK = 3 };

__device__ __forceinline__ unsigned f2key(float x) {
    unsigned u = __float_as_uint(x);
    return (u & 0x80000000u) ? ~u : (u | 0x80000000u);
}
__device__ __forceinline__ float key2f(unsigned k) {
    return (k & 0x80000000u) ? __uint_as_float(k & 0x7FFFFFFFu) : __uint_as_float(~k);
}
__device__ __forceinline__ v16h join8(v8h lo, v8h hi) {
    return __builtin_shufflevector(lo, hi, 0,1,2,3,4,5,6,7,8,9,10,11,12,13,14,15);
}

// ---------------------------------------------------------------- init
__global__ void Kinit(unsigned* __restrict__ maxkey, float* __restrict__ Z,
                      float* __restrict__ S2, float* __restrict__ rnk) {
    int i = blockIdx.x * blockDim.x + threadIdx.x;
    if (i < N_ROWS_A) { maxkey[i] = 0x007FFFFFu; /* key(-inf) */ Z[i] = 0.f; S2[i] = 0.f; }
    if (i < N_ROWS_B) rnk[i] = 0.f;
}

// ---------------------------------------------------------------- f32 -> f16 venue table
__global__ void Kcvt(const float* __restrict__ src, _Float16* __restrict__ dst, int n) {
    int i = blockIdx.x * blockDim.x + threadIdx.x;
    if (i < n) dst[i] = (_Float16)src[i];
}

// ---------------------------------------------------------------- time MLP + venue gather
__global__ __launch_bounds__(256)
void Kmlp(const float* __restrict__ timein, const float* __restrict__ w1,
          const float* __restrict__ b1, const float* __restrict__ w2,
          const float* __restrict__ b2, const float* __restrict__ venue_table,
          const int* __restrict__ venue, float* __restrict__ time_emb,
          float* __restrict__ inputs) {
    __shared__ float hid[32 * 128];
    const int b = blockIdx.x;
    const int tid = threadIdx.x;
    for (int i = tid; i < 32 * 128; i += 256) {
        const int l = i >> 7, h = i & 127;
        float acc = b1[h];
        const float* tt = timein + ((size_t)b * 32 + l) * 8;
        #pragma unroll
        for (int k = 0; k < 8; k++) acc += tt[k] * w1[k * 128 + h];
        hid[i] = fmaxf(acc, 0.f);
    }
    __syncthreads();
    for (int i = tid; i < 32 * 64; i += 256) {
        const int l = i >> 6, d = i & 63;
        float acc = b2[d];
        const float* hh = hid + l * 128;
        #pragma unroll 8
        for (int k = 0; k < 128; k++) acc += hh[k] * w2[k * 64 + d];
        const size_t o = ((size_t)b * 32 + l) * 64 + d;
        time_emb[o] = acc;
        inputs[o] = acc + venue_table[(size_t)venue[b * 32 + l] * 64 + d];
    }
}

// ---------------------------------------------------------------- RNN (serial, one WG)
__global__ __launch_bounds__(512)
void Krnn(const int* __restrict__ user, const float* __restrict__ user_table,
          const float* __restrict__ Wih, const float* __restrict__ Whh,
          const float* __restrict__ bih, const float* __restrict__ bhh,
          const float* __restrict__ inputs, const float* __restrict__ time_emb,
          _Float16* __restrict__ seqA, _Float16* __restrict__ seqB) {
    extern __shared__ char dynsm[];
    float* hA  = (float*)dynsm;            // 64x64
    float* hB  = hA + 4096;                // 64x64
    float* wih = hB + 4096;                // 64x64
    float* whh = wih + 4096;               // 64x64
    const int tid = threadIdx.x;
    for (int i = tid; i < 4096; i += 512) {
        wih[i] = Wih[i]; whh[i] = Whh[i];
        const int b = i >> 6, d = i & 63;
        hA[i] = user_table[(size_t)user[b] * 64 + d];
    }
    __syncthreads();
    float* hc = hA; float* hn = hB;
    for (int t = 0; t < 32; t++) {
        for (int i = tid; i < 4096; i += 512) {
            const int b = i >> 6, d = i & 63;
            const float sv = hc[i] - time_emb[((size_t)b * 32 + t) * 64 + d];
            if (t < 29) seqA[((size_t)b * 29 + t) * 64 + d] = (_Float16)sv;
            else        seqB[((size_t)b * 3 + (t - 29)) * 64 + d] = (_Float16)sv;
        }
        if (t < 31) {
            __syncthreads();
            for (int i = tid; i < 4096; i += 512) {
                const int b = i >> 6, d = i & 63;
                const float* x = inputs + ((size_t)b * 32 + t) * 64;
                const float* hrow = hc + b * 64;
                const float* wi = wih + d * 64;
                const float* wh = whh + d * 64;
                float acc = bih[d] + bhh[d];
                #pragma unroll 8
                for (int k = 0; k < 64; k++) acc += x[k] * wi[k] + hrow[k] * wh[k];
                hn[i] = tanhf(acc);
            }
            __syncthreads();
            float* tp = hc; hc = hn; hn = tp;
        }
    }
}

// ---------------------------------------------------------------- target logits (direct dot)
__global__ void Ktgt(const _Float16* __restrict__ seqA, const _Float16* __restrict__ seqB,
                     const _Float16* __restrict__ venue_h, const int* __restrict__ venue,
                     float* __restrict__ tgtA, float* __restrict__ tgtB) {
    const int r = blockIdx.x * blockDim.x + threadIdx.x;
    if (r < N_ROWS_A) {
        const int b = r / 29, t = r % 29;
        const int tg = venue[b * 32 + t];
        const _Float16* s = seqA + (size_t)r * 64;
        const _Float16* v = venue_h + (size_t)tg * 64;
        float acc = 0.f;
        #pragma unroll 8
        for (int k = 0; k < 64; k++) acc += (float)s[k] * (float)v[k];
        tgtA[r] = acc;
    } else if (r >= 2048 && r < 2048 + N_ROWS_B) {
        const int r2 = r - 2048;
        const int b = r2 / 3, i = r2 % 3;
        const int tg = venue[b * 32 + 29 + i];
        const _Float16* s = seqB + (size_t)r2 * 64;
        const _Float16* v = venue_h + (size_t)tg * 64;
        float acc = 0.f;
        #pragma unroll 8
        for (int k = 0; k < 64; k++) acc += (float)s[k] * (float)v[k];
        tgtB[r2] = acc;
    }
}

// ---------------------------------------------------------------- streaming GEMM passes (WMMA)
template <int MODE>
__global__ __launch_bounds__(128)
void Kpass(const _Float16* __restrict__ seq_h, int nrows,
           const _Float16* __restrict__ venue_h,
           unsigned* __restrict__ maxkey, float* __restrict__ Z,
           float* __restrict__ S2out, const float* __restrict__ tgt,
           float* __restrict__ rankout) {
    extern __shared__ char dynsm[];
    _Float16* ldsv = (_Float16*)dynsm;   // [NV_TILE][64]
    const int vbase = blockIdx.x * NV_TILE;
    const int tid = threadIdx.x;

    // ---- Stage venue slab: async DMA global -> LDS (ASYNCcnt-tracked, no VGPR round trip).
    // OOB rows are clamped to the last valid row; their contribution is masked later.
    {
        const unsigned lds0 = (unsigned)(size_t)(void*)ldsv;  // low 32 bits = LDS offset
        for (int cidx = tid; cidx < NV_TILE * 8; cidx += 128) {
            const int row = cidx >> 3, part = cidx & 7;
            int g = vbase + row;
            if (g >= N_VENUES) g = N_VENUES - 1;
            const unsigned voff = ((unsigned)g * 64u + (unsigned)part * 8u) * 2u;  // bytes
            const unsigned ldsa = lds0 + ((unsigned)(row * 64 + part * 8)) * 2u;   // bytes
            asm volatile("global_load_async_to_lds_b128 %0, %1, %2"
                         :: "v"(ldsa), "v"(voff), "s"(venue_h) : "memory");
        }
        asm volatile("s_wait_asynccnt 0x0" ::: "memory");
    }
    __syncthreads();

    const int wave = tid >> 5;
    const int lane = tid & 31;
    const int hi = lane >> 4;   // lane group: K-half of fragments / row-half of C
    const int ln = lane & 15;
    const int ntiles = nrows >> 4;

    for (int tile = wave; tile < ntiles; tile += 4) {
        const int row0 = tile << 4;
        if (tile + 4 < ntiles)
            __builtin_prefetch(seq_h + (size_t)(row0 + 64) * 64, 0, 1);

        // A fragment: 16x32 f16 layout (lane<16: K 0..7|16..23, lane>=16: K 8..15|24..31)
        const _Float16* arow = seq_h + (size_t)(row0 + ln) * 64;
        const v16h a0 = join8(*(const v8h*)(arow + hi * 8),      *(const v8h*)(arow + 16 + hi * 8));
        const v16h a1 = join8(*(const v8h*)(arow + 32 + hi * 8), *(const v8h*)(arow + 48 + hi * 8));

        // Per-row stats needed by this mode (rows r + 8*hi of this tile)
        float rs0[8], rs1[8];
        if (MODE != MODE_MAX) {
            #pragma unroll
            for (int r = 0; r < 8; r++) {
                const int rowi = row0 + r + 8 * hi;
                if (MODE == MODE_Z) { rs0[r] = key2f(maxkey[rowi]); rs1[r] = 0.f; }
                else if (MODE == MODE_S2) { rs0[r] = key2f(maxkey[rowi]); rs1[r] = 1.0f / Z[rowi]; }
                else { rs0[r] = tgt[rowi]; rs1[r] = 0.f; }
            }
        }

        float acc[8];
        #pragma unroll
        for (int r = 0; r < 8; r++) acc[r] = (MODE == MODE_MAX) ? -3.4e38f : 0.f;

        #pragma unroll 4
        for (int sv = 0; sv < NSUB; ++sv) {
            const int nIdx = vbase + sv * 16 + ln;
            const float vm = (nIdx < N_VENUES) ? 1.f : 0.f;   // branchless validity mask
            // B fragment: 32x16 f16 (lane n=ln holds K = hi*8 + 0..7 | 16 + hi*8 + 0..7 of column n)
            const _Float16* brow = ldsv + (size_t)(sv * 16 + ln) * 64;
            const v16h b0 = join8(*(const v8h*)(brow + hi * 8),      *(const v8h*)(brow + 16 + hi * 8));
            const v16h b1 = join8(*(const v8h*)(brow + 32 + hi * 8), *(const v8h*)(brow + 48 + hi * 8));

            v8f c = {0.f, 0.f, 0.f, 0.f, 0.f, 0.f, 0.f, 0.f};
            c = __builtin_amdgcn_wmma_f32_16x16x32_f16(false, a0, false, b0, (short)0, c, false, false);
            c = __builtin_amdgcn_wmma_f32_16x16x32_f16(false, a1, false, b1, (short)0, c, false, false);

            if (MODE == MODE_MAX) {
                #pragma unroll
                for (int r = 0; r < 8; r++) {
                    const float x = (vm > 0.f) ? c[r] : -3.4e38f;  // v_cndmask
                    acc[r] = fmaxf(acc[r], x);
                }
            } else if (MODE == MODE_Z) {
                #pragma unroll
                for (int r = 0; r < 8; r++) acc[r] += vm * __expf(c[r] - rs0[r]);
            } else if (MODE == MODE_S2) {
                #pragma unroll
                for (int r = 0; r < 8; r++)
                    acc[r] += vm * __expf(__expf(c[r] - rs0[r]) * rs1[r]);
            } else { // MODE_RANK
                #pragma unroll
                for (int r = 0; r < 8; r++) acc[r] += (c[r] > rs0[r]) ? vm : 0.f;
            }
        }

        // Reduce over the 16 venue-columns held by this lane group, then atomics per row.
        #pragma unroll
        for (int r = 0; r < 8; r++) {
            float v = acc[r];
            #pragma unroll
            for (int ofs = 1; ofs < 16; ofs <<= 1) {
                const float o = __shfl_xor(v, ofs, 32);
                v = (MODE == MODE_MAX) ? fmaxf(v, o) : (v + o);
            }
            if (ln == 0) {
                const int rowi = row0 + r + 8 * hi;
                if (MODE == MODE_MAX)      atomicMax(&maxkey[rowi], f2key(v));
                else if (MODE == MODE_Z)   atomicAdd(&Z[rowi], v);
                else if (MODE == MODE_S2)  atomicAdd(&S2out[rowi], v);
                else                       atomicAdd(&rankout[rowi], v);
            }
        }
    }
}

// ---------------------------------------------------------------- final reduce
__global__ __launch_bounds__(256)
void Kfinal(const unsigned* __restrict__ maxkey, const float* __restrict__ Z,
            const float* __restrict__ S2, const float* __restrict__ tgtA,
            const float* __restrict__ rnk, float* __restrict__ out) {
    __shared__ float red[256];
    const int tid = threadIdx.x;
    float s = 0.f;
    for (int r = tid; r < N_ROWS_A; r += 256) {
        const float mx = key2f(maxkey[r]);
        const float pt = __expf(tgtA[r] - mx) / Z[r];   // p[target]
        s += pt - __logf(S2[r]);                        // log_softmax(probs)[target]
    }
    red[tid] = s; __syncthreads();
    for (int ofs = 128; ofs > 0; ofs >>= 1) { if (tid < ofs) red[tid] += red[tid + ofs]; __syncthreads(); }
    if (tid == 0) out[0] = -red[0] / (float)N_ROWS_A;

    float c[4] = {0.f, 0.f, 0.f, 0.f};
    const float ks[4] = {1.f, 5.f, 10.f, 20.f};
    for (int r = tid; r < N_ROWS_B; r += 256) {
        const float rk = rnk[r];   // #venues strictly beating the target
        #pragma unroll
        for (int j = 0; j < 4; j++) c[j] += (rk < ks[j]) ? 1.f : 0.f;
    }
    for (int j = 0; j < 4; j++) {
        __syncthreads(); red[tid] = c[j]; __syncthreads();
        for (int ofs = 128; ofs > 0; ofs >>= 1) { if (tid < ofs) red[tid] += red[tid + ofs]; __syncthreads(); }
        if (tid == 0) out[1 + j] = red[0];
    }
    if (tid == 0) out[5] = (float)N_ROWS_B;
}

// ---------------------------------------------------------------- launch
extern "C" void kernel_launch(void* const* d_in, const int* in_sizes, int n_in,
                              void* d_out, int out_size, void* d_ws, size_t ws_size,
                              hipStream_t stream) {
    (void)in_sizes; (void)n_in; (void)out_size; (void)ws_size;
    const int*   user        = (const int*)  d_in[0];
    const int*   venue       = (const int*)  d_in[1];
    const float* timein      = (const float*)d_in[2];
    const float* venue_table = (const float*)d_in[3];
    const float* user_table  = (const float*)d_in[4];
    const float* w1          = (const float*)d_in[5];
    const float* b1          = (const float*)d_in[6];
    const float* w2          = (const float*)d_in[7];
    const float* b2          = (const float*)d_in[8];
    const float* Wih         = (const float*)d_in[9];
    const float* Whh         = (const float*)d_in[10];
    const float* bih         = (const float*)d_in[11];
    const float* bhh         = (const float*)d_in[12];
    float* out = (float*)d_out;

    char* wp = (char*)d_ws;
    auto alloc = [&](size_t bytes) -> char* {
        char* p = wp; wp += (bytes + 255) & ~(size_t)255; return p;
    };
    _Float16* venue_h = (_Float16*)alloc((size_t)N_VENUES * 64 * 2);
    _Float16* seqA    = (_Float16*)alloc((size_t)N_ROWS_A * 64 * 2);
    _Float16* seqB    = (_Float16*)alloc((size_t)N_ROWS_B * 64 * 2);
    float* time_emb   = (float*)alloc((size_t)64 * 32 * 64 * 4);
    float* inputs     = (float*)alloc((size_t)64 * 32 * 64 * 4);
    unsigned* maxkey  = (unsigned*)alloc((size_t)N_ROWS_A * 4);
    float* Zb         = (float*)alloc((size_t)N_ROWS_A * 4);
    float* S2b        = (float*)alloc((size_t)N_ROWS_A * 4);
    float* tgtA       = (float*)alloc((size_t)N_ROWS_A * 4);
    float* tgtB       = (float*)alloc((size_t)N_ROWS_B * 4);
    float* rnk        = (float*)alloc((size_t)N_ROWS_B * 4);

    Kinit<<<8, 256, 0, stream>>>(maxkey, Zb, S2b, rnk);
    Kcvt<<<(N_VENUES * 64 + 255) / 256, 256, 0, stream>>>(venue_table, venue_h, N_VENUES * 64);
    Kmlp<<<64, 256, 0, stream>>>(timein, w1, b1, w2, b2, venue_table, venue, time_emb, inputs);
    Krnn<<<1, 512, 65536, stream>>>(user, user_table, Wih, Whh, bih, bhh, inputs, time_emb, seqA, seqB);
    Ktgt<<<9, 256, 0, stream>>>(seqA, seqB, venue_h, venue, tgtA, tgtB);

    const int nvb = (N_VENUES + NV_TILE - 1) / NV_TILE;
    const size_t lds = (size_t)NV_TILE * 64 * sizeof(_Float16);
    Kpass<MODE_MAX><<<nvb, 128, lds, stream>>>(seqA, N_ROWS_A, venue_h, maxkey, Zb, S2b, tgtA, rnk);
    Kpass<MODE_Z><<<nvb, 128, lds, stream>>>(seqA, N_ROWS_A, venue_h, maxkey, Zb, S2b, tgtA, rnk);
    Kpass<MODE_S2><<<nvb, 128, lds, stream>>>(seqA, N_ROWS_A, venue_h, maxkey, Zb, S2b, tgtA, rnk);
    Kpass<MODE_RANK><<<nvb, 128, lds, stream>>>(seqB, N_ROWS_B, venue_h, maxkey, Zb, S2b, tgtB, rnk);

    Kfinal<<<1, 256, 0, stream>>>(maxkey, Zb, S2b, tgtA, rnk, out);
}
